// Attention_block_3401614098592
// MI455X (gfx1250) — matrix operand order repaired and checked by reading, not compiled, not run on hardware
//
#include <hip/hip_runtime.h>
#include <cstdint>

// Graph-transformer block for MI455X (gfx1250, wave32, WMMA).
// Pipeline: CSR-build edge aggregation (no float atomics), f16-WMMA GEMMs
// with fp32 accumulation (W staged via gfx1250 async global->LDS DMA),
// wave32 softmax/LN via __shfl_xor reductions.

#define NN   32768      // total nodes (B*P)
#define EE   524288     // edges
#define BBG  128        // graphs
#define PPN  256        // nodes / graph
#define SSQ  32         // seed queries / graph
#define DV   128        // feature dim
#define ROWS_QS 4096    // B*S rows

typedef __attribute__((ext_vector_type(16))) _Float16 v16h;
typedef __attribute__((ext_vector_type(8)))  float    v8f;

__device__ __forceinline__ float warp_sum32(float v) {
#pragma unroll
  for (int off = 16; off > 0; off >>= 1) v += __shfl_xor(v, off, 32);
  return v;
}
__device__ __forceinline__ float warp_max32(float v) {
#pragma unroll
  for (int off = 16; off > 0; off >>= 1) v = fmaxf(v, __shfl_xor(v, off, 32));
  return v;
}
__device__ __forceinline__ void fma4(float4& a, const float4 b, const float c) {
  a.x += b.x * c; a.y += b.y * c; a.z += b.z * c; a.w += b.w * c;
}

// ---------------------------------------------------------------------------
// GEMM: C[M,128] = A[M,128] @ W[128,128] (+ bias).  M multiple of 128.
// Block = 256 threads = 8 waves; wave computes a 16x128 output slab.
// W staged f32 via GLOBAL_LOAD_ASYNC_TO_LDS_B128 (ASYNCcnt), then transposed
// to f16 in LDS; WMMA f32_16x16x32_f16 with fp32 accumulation.
// ---------------------------------------------------------------------------
__global__ __launch_bounds__(256) void k_gemm128(const float* __restrict__ A,
                                                 const float* __restrict__ W,
                                                 const float* __restrict__ bias,
                                                 float* __restrict__ C) {
  __shared__ float    sWf[128 * 128];          // 64 KB raw W (async DMA dest)
  __shared__ _Float16 sWt[128 * 128];          // 32 KB W^T, f16

  // Stage W: 1024 async b128 transfers (per-lane), no VGPR round trip.
  // Generic shared pointers carry the LDS byte offset in their low 32 bits.
  for (int i = threadIdx.x; i < 1024; i += 256) {
    const float*   g = W + (size_t)i * 4;
    const uint32_t l = (uint32_t)(uintptr_t)(const void*)(sWf + (size_t)i * 4);
    asm volatile("global_load_async_to_lds_b128 %0, %1, off"
                 :: "v"(l), "v"(g) : "memory");
  }
  asm volatile("s_wait_asynccnt 0" ::: "memory");
  __syncthreads();

  // One-time transpose + f16 convert (bank conflicts here are amortized away).
  for (int i = threadIdx.x; i < 128 * 128; i += 256) {
    const int k = i >> 7, n = i & 127;         // W row-major: i = k*128 + n
    sWt[n * 128 + k] = (_Float16)sWf[i];
  }
  __syncthreads();

  const int wave  = threadIdx.x >> 5;
  const int lane  = threadIdx.x & 31;
  const int m0    = blockIdx.x * 128 + wave * 16;
  const int hi    = lane >> 4;                 // 0 / 1 (half-wave)
  const int arow  = m0 + (lane & 15);
  const int khiA  = hi << 3;                   // A: hi lanes hold K runs +8
  const int khiB  = hi << 3;                   // B: hi lanes hold K runs +8
  const float* ap = A + (size_t)arow * 128;

  // A fragments for K=128 (4 k-blocks of 32). 16-bit A layout:
  // lane<16: K in {kb..kb+7} U {kb+16..kb+23}; lane>=16: the +8 runs.
  v16h afrag[4];
#pragma unroll
  for (int t4 = 0; t4 < 4; t4++) {
    const int kb = t4 * 32;
    const float4 f0 = *(const float4*)(ap + kb + khiA + 0);
    const float4 f1 = *(const float4*)(ap + kb + khiA + 4);
    const float4 f2 = *(const float4*)(ap + kb + khiA + 16);
    const float4 f3 = *(const float4*)(ap + kb + khiA + 20);
    v16h a;
    a[0]=(_Float16)f0.x; a[1]=(_Float16)f0.y; a[2]=(_Float16)f0.z; a[3]=(_Float16)f0.w;
    a[4]=(_Float16)f1.x; a[5]=(_Float16)f1.y; a[6]=(_Float16)f1.z; a[7]=(_Float16)f1.w;
    a[8]=(_Float16)f2.x; a[9]=(_Float16)f2.y; a[10]=(_Float16)f2.z; a[11]=(_Float16)f2.w;
    a[12]=(_Float16)f3.x; a[13]=(_Float16)f3.y; a[14]=(_Float16)f3.z; a[15]=(_Float16)f3.w;
    afrag[t4] = a;
  }

  const int ccol0 = lane & 15;
  const int rbase = m0 + khiA;                 // C/D: hi lanes hold rows +8

  // Hoist bias (uniform branch; removes per-tile branch + load in store path).
  float bvv[8];
  if (bias) {
#pragma unroll
    for (int nt = 0; nt < 8; nt++) bvv[nt] = bias[nt * 16 + ccol0];
  } else {
#pragma unroll
    for (int nt = 0; nt < 8; nt++) bvv[nt] = 0.0f;
  }

  for (int nt = 0; nt < 8; nt++) {
    const int n0 = nt * 16;
    v8f acc = {};
    const _Float16* wcol = sWt + (size_t)(n0 + ccol0) * 128 + khiB;
#pragma unroll
    for (int t4 = 0; t4 < 4; t4++) {
      v16h bfrag;
#pragma unroll
      for (int j = 0; j < 8; j++) { bfrag[j] = wcol[t4 * 32 + j]; bfrag[8 + j] = wcol[t4 * 32 + 16 + j]; }
      acc = __builtin_amdgcn_wmma_f32_16x16x32_f16(
          /*neg_a=*/false, afrag[t4], /*neg_b=*/false, bfrag,
          /*c_mod=*/(short)0, acc, /*reuse_a=*/false, /*reuse_b=*/false);
    }
    const int ccol = n0 + ccol0;
    const float bv = bvv[nt];
#pragma unroll
    for (int r = 0; r < 8; r++)
      C[(size_t)(rbase + r) * 128 + ccol] = acc[r] + bv;
  }
}

// ---------------------------------------------------------------------------
// CSR construction (count -> scan -> fill), then per-node gather.
// ---------------------------------------------------------------------------
__global__ __launch_bounds__(256) void k_init(int* __restrict__ deg, int* __restrict__ fill) {
  const int i = blockIdx.x * 256 + threadIdx.x;
  if (i < NN) { deg[i] = 0; fill[i] = 0; }
}

__global__ __launch_bounds__(256) void k_count(const int* __restrict__ ei, int* __restrict__ deg) {
  const int e = blockIdx.x * 256 + threadIdx.x;
  if (e < EE) atomicAdd(&deg[ei[EE + e]], 1);   // dst = edge_index[1][e]
}

__global__ __launch_bounds__(256) void k_dinv(const int* __restrict__ deg, float* __restrict__ dinv) {
  const int i = blockIdx.x * 256 + threadIdx.x;
  if (i < NN) dinv[i] = rsqrtf((float)deg[i] + 1.0f);   // +1 self-loop
}

// Exclusive prefix sum of deg[0..NN) -> rowptr. Single block, 1024 threads x 32.
__global__ __launch_bounds__(1024) void k_scan(const int* __restrict__ deg, int* __restrict__ rowptr) {
  __shared__ int ssum[1024];
  const int t = threadIdx.x;
  const int base = t * 32;
  int loc[32];
  int s = 0;
#pragma unroll
  for (int i = 0; i < 32; i++) { loc[i] = deg[base + i]; s += loc[i]; }
  ssum[t] = s;
  __syncthreads();
  for (int off = 1; off < 1024; off <<= 1) {
    const int v = (t >= off) ? ssum[t - off] : 0;
    __syncthreads();
    ssum[t] += v;
    __syncthreads();
  }
  int excl = ssum[t] - s;
#pragma unroll
  for (int i = 0; i < 32; i++) { rowptr[base + i] = excl; excl += loc[i]; }
}

__global__ __launch_bounds__(256) void k_fill(const int* __restrict__ ei,
                                              const int* __restrict__ rowptr,
                                              int* __restrict__ fill,
                                              int* __restrict__ csr_src) {
  const int e = blockIdx.x * 256 + threadIdx.x;
  if (e >= EE) return;
  const int src = ei[e];
  const int dst = ei[EE + e];
  const int pos = rowptr[dst] + atomicAdd(&fill[dst], 1);
  csr_src[pos] = src;
}

// One wave per node: Kout/Vout[n] = sum_e coef*xw[src] + dinv(n)^2*xw[n] + b
__global__ __launch_bounds__(256) void k_gather(const float* __restrict__ xwK,
                                                const float* __restrict__ xwV,
                                                const int* __restrict__ rowptr,
                                                const int* __restrict__ deg,
                                                const int* __restrict__ csr_src,
                                                const float* __restrict__ dinv,
                                                const float* __restrict__ bk,
                                                const float* __restrict__ bv,
                                                float* __restrict__ Kd,
                                                float* __restrict__ Vd) {
  const int n    = blockIdx.x * 8 + (threadIdx.x >> 5);
  const int lane = threadIdx.x & 31;
  const int start = rowptr[n];
  const int cnt   = deg[n];
  const float dn  = dinv[n];
  float4 aK = {0.f, 0.f, 0.f, 0.f};
  float4 aV = {0.f, 0.f, 0.f, 0.f};
  for (int e = 0; e < cnt; e++) {
    const int s = csr_src[start + e];
    if (e + 1 < cnt) {                        // prefetch next neighbor row
      const int s2 = csr_src[start + e + 1];
      __builtin_prefetch(xwK + (size_t)s2 * 128, 0, 0);
      __builtin_prefetch(xwV + (size_t)s2 * 128, 0, 0);
    }
    const float c = dinv[s] * dn;
    fma4(aK, ((const float4*)(xwK + (size_t)s * 128))[lane], c);
    fma4(aV, ((const float4*)(xwV + (size_t)s * 128))[lane], c);
  }
  const float cs = dn * dn;                   // self-loop message
  fma4(aK, ((const float4*)(xwK + (size_t)n * 128))[lane], cs);
  fma4(aV, ((const float4*)(xwV + (size_t)n * 128))[lane], cs);
  const float4 b4k = ((const float4*)bk)[lane];
  const float4 b4v = ((const float4*)bv)[lane];
  aK.x += b4k.x; aK.y += b4k.y; aK.z += b4k.z; aK.w += b4k.w;
  aV.x += b4v.x; aV.y += b4v.y; aV.z += b4v.z; aV.w += b4v.w;
  ((float4*)(Kd + (size_t)n * 128))[lane] = aK;
  ((float4*)(Vd + (size_t)n * 128))[lane] = aV;
}

// ---------------------------------------------------------------------------
// Attention: one block per graph; one wave per (h,s) pair (16 pairs/wave).
// scores -> wave32 softmax -> A@V, Oh = Qh + A@V.
// ---------------------------------------------------------------------------
__global__ __launch_bounds__(256) void k_attn(const float* __restrict__ Qp,
                                              const float* __restrict__ Kd,
                                              const float* __restrict__ Vd,
                                              float* __restrict__ X1) {
  __shared__ float A_lds[8][PPN];
  const int b    = blockIdx.x;
  const int wave = threadIdx.x >> 5;
  const int lane = threadIdx.x & 31;
  const float* Qb = Qp + (size_t)b * SSQ * DV;
  const float* Kb = Kd + (size_t)b * PPN * DV;
  const float* Vb = Vd + (size_t)b * PPN * DV;
  const float scale = 0.08838834764831845f;   // 1/sqrt(DV) = 1/sqrt(128)

  for (int pair = wave; pair < 4 * SSQ; pair += 8) {   // uniform trip count
    const int h = pair >> 5, s = pair & 31;
    const int hoff = h * 32;
    const float q = Qb[s * DV + hoff + lane];          // lane holds q[d=lane]

    float sc[8];
#pragma unroll
    for (int j = 0; j < 8; j++) {
      const int p = lane + 32 * j;
      const float* kp = Kb + (size_t)p * DV + hoff;
      float acc = 0.f;
#pragma unroll
      for (int d = 0; d < 32; d++) acc += __shfl(q, d, 32) * kp[d];
      sc[j] = acc * scale;
    }
    // softmax over P=256 (mask is all-valid)
    float m = sc[0];
#pragma unroll
    for (int j = 1; j < 8; j++) m = fmaxf(m, sc[j]);
    m = warp_max32(m);
    float sum = 0.f;
#pragma unroll
    for (int j = 0; j < 8; j++) { sc[j] = __expf(sc[j] - m); sum += sc[j]; }
    sum = warp_sum32(sum);
    const float inv = 1.0f / sum;
#pragma unroll
    for (int j = 0; j < 8; j++) A_lds[wave][lane + 32 * j] = sc[j] * inv;
    __syncthreads();                                   // uniform across waves

    // lane = output dim d within head
    float o = 0.f;
    for (int p = 0; p < PPN; p++)
      o += A_lds[wave][p] * Vb[(size_t)p * DV + hoff + lane];
    X1[((size_t)b * SSQ + s) * DV + hoff + lane] = q + o;
  }
}

// ---------------------------------------------------------------------------
// LayerNorm over last dim (128): one wave per row (in-place capable).
// ---------------------------------------------------------------------------
__global__ __launch_bounds__(256) void k_ln(const float* __restrict__ X,
                                            const float* __restrict__ g,
                                            const float* __restrict__ bta,
                                            float* __restrict__ Y) {
  const int row  = blockIdx.x * 8 + (threadIdx.x >> 5);
  const int lane = threadIdx.x & 31;
  float4 v = ((const float4*)(X + (size_t)row * DV))[lane];
  const float mu = warp_sum32(v.x + v.y + v.z + v.w) * (1.0f / DV);
  v.x -= mu; v.y -= mu; v.z -= mu; v.w -= mu;
  const float var = warp_sum32(v.x*v.x + v.y*v.y + v.z*v.z + v.w*v.w) * (1.0f / DV);
  const float r = rsqrtf(var + 1e-5f);
  const float4 g4 = ((const float4*)g)[lane];
  const float4 b4 = ((const float4*)bta)[lane];
  float4 o;
  o.x = v.x * r * g4.x + b4.x; o.y = v.y * r * g4.y + b4.y;
  o.z = v.z * r * g4.z + b4.z; o.w = v.w * r * g4.w + b4.w;
  ((float4*)(Y + (size_t)row * DV))[lane] = o;
}

// out = LN1( X1 + relu(X2) )
__global__ __launch_bounds__(256) void k_final(const float* __restrict__ X1,
                                               const float* __restrict__ X2,
                                               const float* __restrict__ g,
                                               const float* __restrict__ bta,
                                               float* __restrict__ out) {
  const int row  = blockIdx.x * 8 + (threadIdx.x >> 5);
  const int lane = threadIdx.x & 31;
  const float4 a = ((const float4*)(X1 + (size_t)row * DV))[lane];
  const float4 c = ((const float4*)(X2 + (size_t)row * DV))[lane];
  float4 v;
  v.x = a.x + fmaxf(c.x, 0.f); v.y = a.y + fmaxf(c.y, 0.f);
  v.z = a.z + fmaxf(c.z, 0.f); v.w = a.w + fmaxf(c.w, 0.f);
  const float mu = warp_sum32(v.x + v.y + v.z + v.w) * (1.0f / DV);
  v.x -= mu; v.y -= mu; v.z -= mu; v.w -= mu;
  const float var = warp_sum32(v.x*v.x + v.y*v.y + v.z*v.z + v.w*v.w) * (1.0f / DV);
  const float r = rsqrtf(var + 1e-5f);
  const float4 g4 = ((const float4*)g)[lane];
  const float4 b4 = ((const float4*)bta)[lane];
  float4 o;
  o.x = v.x * r * g4.x + b4.x; o.y = v.y * r * g4.y + b4.y;
  o.z = v.z * r * g4.z + b4.z; o.w = v.w * r * g4.w + b4.w;
  ((float4*)(out + (size_t)row * DV))[lane] = o;
}

// ---------------------------------------------------------------------------
extern "C" void kernel_launch(void* const* d_in, const int* in_sizes, int n_in,
                              void* d_out, int out_size, void* d_ws, size_t ws_size,
                              hipStream_t stream) {
  const float* Q  = (const float*)d_in[0];   // [B,S,128] -> 4096x128
  const float* x  = (const float*)d_in[1];   // [N,128]
  const int*   ei = (const int*)d_in[2];     // [2,E]
  // d_in[3] = batch (unused: equal-size graphs)
  const float* Wq = (const float*)d_in[4];
  const float* bq = (const float*)d_in[5];
  const float* Wk = (const float*)d_in[6];
  const float* bk = (const float*)d_in[7];
  const float* Wv = (const float*)d_in[8];
  const float* bv = (const float*)d_in[9];
  const float* Wo = (const float*)d_in[10];
  const float* bo = (const float*)d_in[11];
  const float* g0 = (const float*)d_in[12];
  const float* b0 = (const float*)d_in[13];
  const float* g1 = (const float*)d_in[14];
  const float* b1 = (const float*)d_in[15];
  float* out = (float*)d_out;

  char* ws = (char*)d_ws;
  size_t off = 0;
  auto alloc = [&](size_t bytes) -> void* {
    void* p = ws + off;
    off += (bytes + 255) & ~(size_t)255;
    return p;
  };
  int*   deg     = (int*)  alloc((size_t)NN * 4);
  int*   fill    = (int*)  alloc((size_t)NN * 4);
  int*   rowptr  = (int*)  alloc((size_t)NN * 4);
  int*   csr_src = (int*)  alloc((size_t)EE * 4);
  float* dinv    = (float*)alloc((size_t)NN * 4);
  float* xwK     = (float*)alloc((size_t)NN * DV * 4);
  float* xwV     = (float*)alloc((size_t)NN * DV * 4);
  float* Kd      = (float*)alloc((size_t)NN * DV * 4);
  float* Vd      = (float*)alloc((size_t)NN * DV * 4);
  float* Qp      = (float*)alloc((size_t)ROWS_QS * DV * 4);
  float* X1      = (float*)alloc((size_t)ROWS_QS * DV * 4);
  float* X2      = (float*)alloc((size_t)ROWS_QS * DV * 4);

  // 1) degree / CSR build
  k_init <<<(NN + 255) / 256, 256, 0, stream>>>(deg, fill);
  k_count<<<(EE + 255) / 256, 256, 0, stream>>>(ei, deg);
  k_dinv <<<(NN + 255) / 256, 256, 0, stream>>>(deg, dinv);
  k_scan <<<1, 1024, 0, stream>>>(deg, rowptr);
  k_fill <<<(EE + 255) / 256, 256, 0, stream>>>(ei, rowptr, fill, csr_src);

  // 2) dense projections on WMMA
  k_gemm128<<<NN / 128, 256, 0, stream>>>(x, Wk, nullptr, xwK);
  k_gemm128<<<NN / 128, 256, 0, stream>>>(x, Wv, nullptr, xwV);
  k_gemm128<<<ROWS_QS / 128, 256, 0, stream>>>(Q, Wq, bq, Qp);

  // 3) GCN aggregation (gather, no float atomics)
  k_gather<<<NN / 8, 256, 0, stream>>>(xwK, xwV, rowptr, deg, csr_src, dinv,
                                       bk, bv, Kd, Vd);

  // 4) attention + residual
  k_attn<<<BBG, 256, 0, stream>>>(Qp, Kd, Vd, X1);

  // 5) LN0 (in place), FFN GEMM, final LN1 -> out
  k_ln  <<<ROWS_QS / 8, 256, 0, stream>>>(X1, g0, b0, X1);
  k_gemm128<<<ROWS_QS / 128, 256, 0, stream>>>(X1, Wo, bo, X2);
  k_final<<<ROWS_QS / 8, 256, 0, stream>>>(X1, X2, g1, b1, out);
}
